// GNNModel_11441792877423
// MI455X (gfx1250) — matrix operand order repaired and checked by reading, not compiled, not run on hardware
//
#include <hip/hip_runtime.h>
#include <stdint.h>
#include <stddef.h>

// ---------------- problem constants ----------------
#define NN    50000
#define EE    400000
#define INF_  300
#define HIDD  512
#define HEADS 4
#define DH    128
#define CCLS  23

typedef __attribute__((ext_vector_type(16))) _Float16 v16h;
typedef __attribute__((ext_vector_type(8)))  _Float16 v8h;
typedef __attribute__((ext_vector_type(8)))  float    v8f;

// ---------------- small device helpers ----------------
__device__ __forceinline__ float waveReduceSum(float v) {
#pragma unroll
  for (int m = 16; m >= 1; m >>= 1) v += __shfl_xor(v, m, 32);
  return v;
}

// blockDim.x == 256 (8 waves). tmp must hold >= 8 floats.
__device__ __forceinline__ float blockReduceSum(float v, float* tmp) {
  const int lane = threadIdx.x & 31;
  const int wv   = threadIdx.x >> 5;
  v = waveReduceSum(v);
  if (lane == 0) tmp[wv] = v;
  __syncthreads();
  float r = (threadIdx.x < 8) ? tmp[threadIdx.x] : 0.0f;
  if (wv == 0) {
#pragma unroll
    for (int m = 4; m >= 1; m >>= 1) r += __shfl_xor(r, m, 32);
    if (lane == 0) tmp[0] = r;
  }
  __syncthreads();
  float out = tmp[0];
  __syncthreads();
  return out;
}

__device__ __forceinline__ void atomicMaxFloat(float* addr, float val) {
  int old = __float_as_int(*addr);
  while (__int_as_float(old) < val) {
    int assumed = old;
    old = atomicCAS((int*)addr, assumed, __float_as_int(val));
    if (old == assumed) break;
  }
}

// ---------------- kernel 1: build x0 = [emb[label], feats] as f16, K padded 600->608 ----
__global__ __launch_bounds__(256) void build_x0_kernel(
    const float* __restrict__ feats, const int* __restrict__ labels,
    const float* __restrict__ emb, _Float16* __restrict__ x0) {
  const long total = (long)NN * 608;
  for (long idx = (long)blockIdx.x * blockDim.x + threadIdx.x; idx < total;
       idx += (long)gridDim.x * blockDim.x) {
    long row = idx / 608;
    int  c   = (int)(idx - row * 608);
    float v;
    if (c < INF_)      v = emb[(long)labels[row] * INF_ + c];
    else if (c < 600)  v = feats[row * INF_ + (c - INF_)];
    else               v = 0.0f;
    x0[idx] = (_Float16)v;
  }
}

// ---------------- kernel 2: weight f32 [K,Nc] -> f16 transposed [NcPad, Kp] ------------
__global__ __launch_bounds__(256) void convert_wt_kernel(
    const float* __restrict__ W, _Float16* __restrict__ Bt,
    int K, int Nc, int Kp, int NcPad) {
  const long total = (long)NcPad * Kp;
  for (long idx = (long)blockIdx.x * blockDim.x + threadIdx.x; idx < total;
       idx += (long)gridDim.x * blockDim.x) {
    long n = idx / Kp;
    int  k = (int)(idx - n * Kp);
    float v = (k < K && n < Nc) ? W[(long)k * Nc + n] : 0.0f;
    Bt[idx] = (_Float16)v;
  }
}

// ---------------- kernel 3: WMMA GEMM ---------------------------------------------------
// C[M, ldc] = A[M, Kp] (f16 row-major) * Bt[NcPad, Kp]^T  (+ bias).
// Block: 256 threads = 8 waves; block computes 64 rows x 512 cols.
// Each wave: 4 M-tiles x 4 N-tiles (16x16 each) -> 16 WMMA accumulators (128 VGPRs).
// Last block clamps A-row reads to M-1 and masks stores (M need not be mult of 64).
__global__ __launch_bounds__(256) void gemm_wmma_kernel(
    const _Float16* __restrict__ A, int lda,
    const _Float16* __restrict__ Bt, int ldb,
    float* __restrict__ C, int ldc,
    const float* __restrict__ bias,
    int Kp, int ncolStore, int M) {
  const int lane = threadIdx.x & 31;
  const int wv   = threadIdx.x >> 5;
  const int l15  = lane & 15;
  const int hi   = lane >> 4;  // 0 or 1 (K-half select per ISA layout)

  const long rowBase = (long)blockIdx.x * 64;
  const _Float16* aP[4];
#pragma unroll
  for (int mt = 0; mt < 4; mt++) {
    long r = rowBase + mt * 16 + l15;
    if (r > (long)M - 1) r = (long)M - 1;  // clamp; stores masked below
    aP[mt] = A + r * lda + (hi ? 8 : 0);
  }
  const int colBase = wv * 64;
  const int bsel = hi ? 8 : 0;
  const _Float16* bP[4];
#pragma unroll
  for (int nt = 0; nt < 4; nt++)
    bP[nt] = Bt + (long)(colBase + nt * 16 + l15) * ldb + bsel;

  v8f acc[4][4];
#pragma unroll
  for (int mt = 0; mt < 4; mt++)
#pragma unroll
    for (int nt = 0; nt < 4; nt++) acc[mt][nt] = (v8f){};

  for (int kb = 0; kb < Kp; kb += 32) {
    v16h a[4];
#pragma unroll
    for (int mt = 0; mt < 4; mt++) {
      __builtin_prefetch(aP[mt] + kb + 256, 0, 1);  // global_prefetch_b8 (speculative)
      v8h lo  = *(const v8h*)(aP[mt] + kb);
      v8h hi8 = *(const v8h*)(aP[mt] + kb + 16);
      a[mt] = __builtin_shufflevector(lo, hi8, 0,1,2,3,4,5,6,7,8,9,10,11,12,13,14,15);
    }
#pragma unroll
    for (int nt = 0; nt < 4; nt++) {
      v8h t0 = *(const v8h*)(bP[nt] + kb);
      v8h t1 = *(const v8h*)(bP[nt] + kb + 16);
      v16h b = __builtin_shufflevector(t0, t1, 0,1,2,3,4,5,6,7,8,9,10,11,12,13,14,15);
#pragma unroll
      for (int mt = 0; mt < 4; mt++)
        acc[mt][nt] = __builtin_amdgcn_wmma_f32_16x16x32_f16(
            false, a[mt], false, b, (short)0, acc[mt][nt], false, false);
    }
  }

  // store (C VGPR layout: elem r -> row rBase + r, col = tile + lane&15; hi adds 8 rows)
#pragma unroll
  for (int mt = 0; mt < 4; mt++) {
    const long rB = rowBase + mt * 16 + (hi ? 8 : 0);
#pragma unroll
    for (int nt = 0; nt < 4; nt++) {
      const int c = colBase + nt * 16 + l15;
      if (c < ncolStore) {
        const float bv = bias ? bias[c] : 0.0f;
#pragma unroll
        for (int r = 0; r < 8; r++) {
          const long rowC = rB + r;
          if (rowC < M) C[rowC * ldc + c] = acc[mt][nt][r] + bv;
        }
      }
    }
  }
}

// ---------------- kernel 4: row-wise bias + (optional LN) + activation -----------------
// act: 0=none, 1=relu, 2=elu. One block (256 thr) per row; F <= 512.
__global__ __launch_bounds__(256) void epilogue_kernel(
    const float* __restrict__ G, const float* __restrict__ bias,
    const float* __restrict__ lnw, const float* __restrict__ lnb,
    int F, int act,
    float* __restrict__ outF32, _Float16* __restrict__ outF16, int Fp) {
  __shared__ float sbuf[512];
  __shared__ float tmp[8];
  const long row = blockIdx.x;
  const float* g = G + row * F;
  float s = 0.0f, ss = 0.0f;
  for (int c = threadIdx.x; c < F; c += blockDim.x) {
    float x = g[c] + (bias ? bias[c] : 0.0f);
    sbuf[c] = x;
    s += x; ss += x * x;
  }
  float u = 0.0f, rinv = 1.0f;
  if (lnw) {
    float ts  = blockReduceSum(s, tmp);
    float tss = blockReduceSum(ss, tmp);
    u = ts / (float)F;
    float var = tss / (float)F - u * u;
    rinv = rsqrtf(var + 1e-12f);
  }
  for (int c = threadIdx.x; c < F; c += blockDim.x) {
    float x = sbuf[c];
    if (lnw) x = lnw[c] * (x - u) * rinv + lnb[c];
    if (act == 1)      x = fmaxf(x, 0.0f);
    else if (act == 2) x = (x > 0.0f) ? x : (__expf(x) - 1.0f);
    if (outF32) outF32[row * F + c] = x;
    if (outF16) outF16[row * Fp + c] = (_Float16)x;
  }
  if (outF16) {
    for (int c = F + threadIdx.x; c < Fp; c += blockDim.x) outF16[row * Fp + c] = (_Float16)0.0f;
  }
}

// ---------------- kernel 5: f32 [N,F] -> f16 [N,Fp] with zero pad ----------------------
__global__ __launch_bounds__(256) void f32_to_f16_pad_kernel(
    const float* __restrict__ src, int F, _Float16* __restrict__ dst, int Fp) {
  const long total = (long)NN * Fp;
  for (long idx = (long)blockIdx.x * blockDim.x + threadIdx.x; idx < total;
       idx += (long)gridDim.x * blockDim.x) {
    long row = idx / Fp;
    int  c   = (int)(idx - row * Fp);
    dst[idx] = (c < F) ? (_Float16)src[row * F + c] : (_Float16)0.0f;
  }
}

// ---------------- kernel 6: per-(node,head) attention dots el/er -----------------------
__global__ __launch_bounds__(256) void el_er_kernel(
    const float* __restrict__ f, const float* __restrict__ attn_l,
    const float* __restrict__ attn_r, float* __restrict__ el, float* __restrict__ er) {
  const long gw = ((long)blockIdx.x * blockDim.x + threadIdx.x) >> 5;
  const int lane = threadIdx.x & 31;
  if (gw >= (long)NN * HEADS) return;
  const long i = gw >> 2;
  const int  h = (int)(gw & 3);
  const float* fp = f + i * HIDD + h * DH;
  const float* al = attn_l + h * DH;
  const float* ar = attn_r + h * DH;
  float sl = 0.0f, sr = 0.0f;
  for (int d = lane; d < DH; d += 32) {
    float v = fp[d];
    sl += v * al[d];
    sr += v * ar[d];
  }
  sl = waveReduceSum(sl);
  sr = waveReduceSum(sr);
  if (lane == 0) { el[gw] = sl; er[gw] = sr; }
}

// ---------------- kernel 7: init GAT accumulators --------------------------------------
__global__ __launch_bounds__(256) void gat_init_kernel(
    float* __restrict__ acc, float* __restrict__ mmax, float* __restrict__ ssum) {
  const long total = (long)NN * HIDD;
  for (long idx = (long)blockIdx.x * blockDim.x + threadIdx.x; idx < total;
       idx += (long)gridDim.x * blockDim.x) {
    acc[idx] = 0.0f;
    if (idx < (long)NN * HEADS) { mmax[idx] = -3.0e38f; ssum[idx] = 0.0f; }
  }
}

// ---------------- kernel 8: edge scores + segment max ----------------------------------
__global__ __launch_bounds__(256) void edge_score_kernel(
    const int* __restrict__ src, const int* __restrict__ dst,
    const float* __restrict__ el, const float* __restrict__ er,
    float* __restrict__ eA, float* __restrict__ mmax) {
  const long total = (long)EE * HEADS;
  for (long idx = (long)blockIdx.x * blockDim.x + threadIdx.x; idx < total;
       idx += (long)gridDim.x * blockDim.x) {
    long e = idx >> 2;
    int  h = (int)(idx & 3);
    float v = el[(long)src[e] * HEADS + h] + er[(long)dst[e] * HEADS + h];
    v = (v > 0.0f) ? v : 0.2f * v;  // leaky_relu 0.2
    eA[idx] = v;
    atomicMaxFloat(&mmax[(long)dst[e] * HEADS + h], v);
  }
}

// ---------------- kernel 9: exp + segment sum ------------------------------------------
__global__ __launch_bounds__(256) void edge_exp_kernel(
    const int* __restrict__ dst, float* __restrict__ eA,
    const float* __restrict__ mmax, float* __restrict__ ssum) {
  const long total = (long)EE * HEADS;
  for (long idx = (long)blockIdx.x * blockDim.x + threadIdx.x; idx < total;
       idx += (long)gridDim.x * blockDim.x) {
    long e = idx >> 2;
    int  h = (int)(idx & 3);
    float a = __expf(eA[idx] - mmax[(long)dst[e] * HEADS + h]);
    eA[idx] = a;
    atomicAdd(&ssum[(long)dst[e] * HEADS + h], a);
  }
}

// ---------------- kernel 10: weighted scatter-add aggregation --------------------------
// One block (128 thr) per edge; thread t handles 4 consecutive channels.
__global__ __launch_bounds__(128) void edge_agg_kernel(
    const int* __restrict__ src, const int* __restrict__ dst,
    const float* __restrict__ eA, const float* __restrict__ ssum,
    const float* __restrict__ f, float* __restrict__ out) {
  const long e = blockIdx.x;
  const int c0 = threadIdx.x * 4;
  const int h  = c0 >> 7;
  const long s = src[e], d = dst[e];
  const float w = eA[e * HEADS + h] / ssum[d * HEADS + h];
  const float4 fv = *(const float4*)(f + s * HIDD + c0);
  atomicAdd(&out[d * HIDD + c0 + 0], w * fv.x);
  atomicAdd(&out[d * HIDD + c0 + 1], w * fv.y);
  atomicAdd(&out[d * HIDD + c0 + 2], w * fv.z);
  atomicAdd(&out[d * HIDD + c0 + 3], w * fv.w);
}

// ---------------- kernel 11: 2-way softmax combine + LN -> f16 -------------------------
__global__ __launch_bounds__(256) void combine_kernel(
    const float* __restrict__ h1, const float* __restrict__ h4,
    const float* __restrict__ attn_w, const float* __restrict__ attn_b,
    const float* __restrict__ lnw, const float* __restrict__ lnb,
    _Float16* __restrict__ hOut) {
  __shared__ float sv[512];
  __shared__ float tmp[8];
  const long row = blockIdx.x;
  const float* p1 = h1 + row * HIDD;
  const float* p4 = h4 + row * HIDD;
  float s1 = 0.0f, s4 = 0.0f;
  for (int c = threadIdx.x; c < HIDD; c += blockDim.x) {
    float w = attn_w[c];
    s1 += p1[c] * w;
    s4 += p4[c] * w;
  }
  s1 = blockReduceSum(s1, tmp) + attn_b[0];
  s4 = blockReduceSum(s4, tmp) + attn_b[0];
  float m  = fmaxf(s1, s4);
  float e1 = __expf(s1 - m), e4 = __expf(s4 - m);
  float inv = 1.0f / (e1 + e4);
  float w1 = e1 * inv, w4 = e4 * inv;

  float s = 0.0f, ss = 0.0f;
  for (int c = threadIdx.x; c < HIDD; c += blockDim.x) {
    float v = w1 * p1[c] + w4 * p4[c];
    sv[c] = v;
    s += v; ss += v * v;
  }
  float ts  = blockReduceSum(s, tmp);
  float tss = blockReduceSum(ss, tmp);
  float u   = ts / (float)HIDD;
  float var = tss / (float)HIDD - u * u;
  float rinv = rsqrtf(var + 1e-12f);
  for (int c = threadIdx.x; c < HIDD; c += blockDim.x) {
    float y = lnw[c] * (sv[c] - u) * rinv + lnb[c];
    hOut[row * HIDD + c] = (_Float16)y;
  }
}

// =======================================================================================
extern "C" void kernel_launch(void* const* d_in, const int* in_sizes, int n_in,
                              void* d_out, int out_size, void* d_ws, size_t ws_size,
                              hipStream_t stream) {
  (void)in_sizes; (void)n_in; (void)out_size; (void)ws_size;

  // -------- inputs (setup_inputs dict order; params as sorted pytree) --------
  const float* feats  = (const float*)d_in[0];
  const int*   labels = (const int*)  d_in[1];
  const int*   src    = (const int*)  d_in[2];
  const int*   dst    = (const int*)  d_in[3];
  const float* emb    = (const float*)d_in[4];
  // fmlp: b1,b2,lnb,lnw,w1,w2
  const float* f_b1  = (const float*)d_in[5];
  const float* f_b2  = (const float*)d_in[6];
  const float* f_lnb = (const float*)d_in[7];
  const float* f_lnw = (const float*)d_in[8];
  const float* f_w1  = (const float*)d_in[9];   // [600,512]
  const float* f_w2  = (const float*)d_in[10];  // [512,300]
  // head: b1,b2,lnb,lnw,w1,w2
  const float* h_b1  = (const float*)d_in[11];
  const float* h_b2  = (const float*)d_in[12];
  const float* h_lnb = (const float*)d_in[13];
  const float* h_lnw = (const float*)d_in[14];
  const float* h_w1  = (const float*)d_in[15];  // [512,512]
  const float* h_w2  = (const float*)d_in[16];  // [512,23]
  // layers[l]: attn_b,attn_l,attn_r,attn_w,gat_b,gat_w,ln_gat_b,ln_gat_w,
  //            ln_out_b,ln_out_w,ln_skip_b,ln_skip_w,skip_b,skip_w
  auto L = [&](int l, int k) { return (const float*)d_in[17 + 14 * l + k]; };

  // -------- workspace carve --------
  uintptr_t p = (uintptr_t)d_ws;
  auto carve = [&](size_t bytes) { uintptr_t r = p; p += (bytes + 255) & ~(size_t)255; return (void*)r; };
  _Float16* Xf16 = (_Float16*)carve((size_t)NN * 640 * 2);   // activations (K padded)
  _Float16* Yf16 = (_Float16*)carve((size_t)NN * 512 * 2);   // second f16 buffer
  _Float16* Wf16 = (_Float16*)carve((size_t)512 * 640 * 2);  // transposed f16 weights
  float* F32a = (float*)carve((size_t)NN * 512 * 4);
  float* F32b = (float*)carve((size_t)NN * 512 * 4);
  float* F32c = (float*)carve((size_t)NN * 512 * 4);
  float* elb  = (float*)carve((size_t)NN * HEADS * 4);
  float* erb  = (float*)carve((size_t)NN * HEADS * 4);
  float* mmax = (float*)carve((size_t)NN * HEADS * 4);
  float* ssum = (float*)carve((size_t)NN * HEADS * 4);
  float* eA   = (float*)carve((size_t)EE * HEADS * 4);

  auto cdiv = [](long a, long b) { return (int)((a + b - 1) / b); };
  const int GEMM_GRID = cdiv(NN, 64);  // 782 blocks of 64 rows (last block masked)

  auto convW = [&](const float* W, int K, int Nc, int Kp) {
    convert_wt_kernel<<<cdiv((long)512 * Kp, 256), 256, 0, stream>>>(W, Wf16, K, Nc, Kp, 512);
  };
  auto gemm = [&](const _Float16* A, int lda, float* C, int ldc, const float* bias,
                  int Kp, int ncolStore) {
    gemm_wmma_kernel<<<GEMM_GRID, 256, 0, stream>>>(A, lda, Wf16, Kp, C, ldc, bias, Kp,
                                                    ncolStore, NN);
  };

  // ================= feature MLP: concat -> 512 (LN,ReLU) -> 300 =================
  build_x0_kernel<<<cdiv((long)NN * 608, 256), 256, 0, stream>>>(feats, labels, emb, Xf16);
  convW(f_w1, 600, 512, 608);
  gemm(Xf16, 608, F32a, 512, nullptr, 608, 512);
  epilogue_kernel<<<NN, 256, 0, stream>>>(F32a, f_b1, f_lnw, f_lnb, 512, /*relu*/1,
                                          nullptr, Yf16, 512);
  convW(f_w2, 512, 300, 512);
  gemm(Yf16, 512, F32b, 300, f_b2, 512, 300);
  f32_to_f16_pad_kernel<<<cdiv((long)NN * 320, 256), 256, 0, stream>>>(F32b, 300, Xf16, 320);

  // ================= GNN layers =================
  for (int l = 0; l < 2; l++) {
    const int din  = (l == 0) ? 300 : 512;
    const int dinp = (l == 0) ? 320 : 512;
    const float* attn_b   = L(l, 0);
    const float* attn_l   = L(l, 1);
    const float* attn_r   = L(l, 2);
    const float* attn_w   = L(l, 3);
    const float* gat_b    = L(l, 4);
    const float* gat_w    = L(l, 5);
    const float* ln_gat_b = L(l, 6);
    const float* ln_gat_w = L(l, 7);
    const float* ln_out_b = L(l, 8);
    const float* ln_out_w = L(l, 9);
    const float* ln_skp_b = L(l, 10);
    const float* ln_skp_w = L(l, 11);
    const float* skip_b   = L(l, 12);
    const float* skip_w   = L(l, 13);

    // skip branch: h1 = elu(LN(h @ skip_w + skip_b))   -> F32a
    convW(skip_w, din, 512, dinp);
    gemm(Xf16, dinp, F32a, 512, nullptr, dinp, 512);
    epilogue_kernel<<<NN, 256, 0, stream>>>(F32a, skip_b, ln_skp_w, ln_skp_b, 512, /*elu*/2,
                                            F32a, nullptr, 0);

    // GAT projection: f = h @ gat_w                    -> F32b
    convW(gat_w, din, 512, dinp);
    gemm(Xf16, dinp, F32b, 512, nullptr, dinp, 512);

    // edge softmax + aggregation                       -> F32c
    el_er_kernel<<<cdiv((long)NN * HEADS * 32, 256), 256, 0, stream>>>(F32b, attn_l, attn_r, elb, erb);
    gat_init_kernel<<<cdiv((long)NN * 512, 256), 256, 0, stream>>>(F32c, mmax, ssum);
    edge_score_kernel<<<cdiv((long)EE * HEADS, 256), 256, 0, stream>>>(src, dst, elb, erb, eA, mmax);
    edge_exp_kernel<<<cdiv((long)EE * HEADS, 256), 256, 0, stream>>>(dst, eA, mmax, ssum);
    edge_agg_kernel<<<EE, 128, 0, stream>>>(src, dst, eA, ssum, F32b, F32c);

    // h4 = elu(LN(gat_out + gat_b))                    -> F32c (in-place)
    epilogue_kernel<<<NN, 256, 0, stream>>>(F32c, gat_b, ln_gat_w, ln_gat_b, 512, /*elu*/2,
                                            F32c, nullptr, 0);

    // softmax-combine + LN -> new h (f16)              -> Xf16 [N,512]
    combine_kernel<<<NN, 256, 0, stream>>>(F32a, F32c, attn_w, attn_b, ln_out_w, ln_out_b, Xf16);
  }

  // ================= head MLP: 512 (LN,ReLU) -> 23 =================
  convW(h_w1, 512, 512, 512);
  gemm(Xf16, 512, F32a, 512, nullptr, 512, 512);
  epilogue_kernel<<<NN, 256, 0, stream>>>(F32a, h_b1, h_lnw, h_lnb, 512, /*relu*/1,
                                          nullptr, Yf16, 512);
  convW(h_w2, 512, 23, 512);
  gemm(Yf16, 512, (float*)d_out, CCLS, h_b2, 512, CCLS);
}
